// LongformerForSequenceClassification_31447750541633
// MI455X (gfx1250) — matrix operand order repaired and checked by reading, not compiled, not run on hardware
//
#include <hip/hip_runtime.h>

// ---------------------------------------------------------------------------
// Longformer encoder (B=2, S=4096, D=768, H=12, Dh=64, F=3072, L=2, w=256)
// bf16 WMMA GEMMs (pre-transposed bf16 weights, double-buffered LDS with
// async global->LDS staging) + flash-style banded attention on gfx1250.
// ---------------------------------------------------------------------------

typedef __bf16 bf16;
typedef __attribute__((ext_vector_type(16))) __bf16 v16bf;
typedef __attribute__((ext_vector_type(8)))  float  v8f;
typedef int v4i_g __attribute__((vector_size(16)));   // matches builtin proto

union F8 { v8f v; float f[8]; };

// ---------------- gfx1250 async global->LDS (guarded) ----------------------
#if __has_builtin(__builtin_amdgcn_global_load_async_to_lds_b128)
#define HAVE_ASYNC_LDS 1
#else
#define HAVE_ASYNC_LDS 0
#endif

// Copy 16 bytes (per lane) global -> LDS. Async DMA when available.
__device__ __forceinline__ void async_copy16(const void* gsrc, void* ldst) {
#if HAVE_ASYNC_LDS
  __builtin_amdgcn_global_load_async_to_lds_b128((v4i_g*)gsrc, (v4i_g*)ldst, 0, 0);
#else
  *reinterpret_cast<uint4*>(ldst) = *reinterpret_cast<const uint4*>(gsrc);
#endif
}

__device__ __forceinline__ void async_wait_all() {
#if HAVE_ASYNC_LDS
#if __has_builtin(__builtin_amdgcn_s_wait_asynccnt)
  __builtin_amdgcn_s_wait_asynccnt(0);
#else
  asm volatile("s_wait_asynccnt 0" ::: "memory");
#endif
#endif
}

// ---------------- WMMA fragment helpers ------------------------------------
__device__ __forceinline__ v8f wmma_bf16(v16bf a, v16bf b, v8f c) {
  return __builtin_amdgcn_wmma_f32_16x16x32_bf16(false, a, false, b,
                                                 (short)0, c, false, false);
}

// A fragment (16x32 bf16), row-major source, leading dim ld.
// lane m=lane&15 holds row m; elems i<8 -> k=hi*8+i, i>=8 -> k=16+hi*8+(i-8).
__device__ __forceinline__ v16bf load_a_frag(const bf16* p, int ld) {
  int lane = threadIdx.x & 31;
  int m = lane & 15, hi = lane >> 4;
  const bf16* row = p + m * ld;
  bf16 e[16];
#pragma unroll
  for (int i = 0; i < 8; ++i) e[i] = row[hi * 8 + i];
#pragma unroll
  for (int i = 0; i < 8; ++i) e[8 + i] = row[16 + hi * 8 + i];
  v16bf v; __builtin_memcpy(&v, e, sizeof(v)); return v;
}

// B fragment (32x16 bf16) from K-transposed storage pt[n*ldk + k] = B[k][n].
// lane n=lane&15 holds col n; elems i<8 -> k=hi*8+i, i>=8 -> k=16+hi*8+(i-8).
__device__ __forceinline__ v16bf load_bt_frag(const bf16* pt, int ldk) {
  int lane = threadIdx.x & 31;
  int n = lane & 15, hi = lane >> 4;
  const bf16* col = pt + n * ldk;
  bf16 e[16];
#pragma unroll
  for (int i = 0; i < 8; ++i) e[i] = col[hi * 8 + i];
#pragma unroll
  for (int i = 0; i < 8; ++i) e[8 + i] = col[16 + hi * 8 + i];
  v16bf v; __builtin_memcpy(&v, e, sizeof(v)); return v;
}

__device__ __forceinline__ float block_reduce_sum(float v, float* red) {
  int t = threadIdx.x;
  red[t] = v; __syncthreads();
  for (int o = blockDim.x >> 1; o > 0; o >>= 1) {
    if (t < o) red[t] += red[t + o];
    __syncthreads();
  }
  float r = red[0]; __syncthreads();
  return r;
}

__device__ __forceinline__ float gelu_tanh(float x) {
  float x3 = x * x * x;
  return 0.5f * x * (1.0f + tanhf(0.7978845608028654f * (x + 0.044715f * x3)));
}

// ---------------------------------------------------------------------------
// Weight prep: f32 [K][N] -> bf16 [N][K] (32x32 LDS tile transpose).
// grid = (N/32, K/32), block = 256 (32x8)
// ---------------------------------------------------------------------------
__global__ __launch_bounds__(256)
void cvt_transpose_kernel(const float* __restrict__ src, bf16* __restrict__ dst,
                          int K, int N) {
  __shared__ float tile[32][33];
  int tx = threadIdx.x & 31, ty = threadIdx.x >> 5;   // 32 x 8
  int n0 = blockIdx.x * 32, k0 = blockIdx.y * 32;
#pragma unroll
  for (int i = 0; i < 4; ++i) {
    int kk = ty + i * 8;
    tile[kk][tx] = src[(size_t)(k0 + kk) * N + n0 + tx];
  }
  __syncthreads();
#pragma unroll
  for (int i = 0; i < 4; ++i) {
    int nn = ty + i * 8;
    dst[(size_t)(n0 + nn) * K + k0 + tx] = (bf16)tile[tx][nn];
  }
}

// ---------------------------------------------------------------------------
// Embedding gather + LayerNorm (D = 768, block = 256, 3 elems/thread)
// ---------------------------------------------------------------------------
__global__ __launch_bounds__(256)
void embed_ln_kernel(const int* __restrict__ ids,
                     const float* __restrict__ wemb,
                     const float* __restrict__ pemb,
                     const float* __restrict__ g, const float* __restrict__ b,
                     float* __restrict__ Xf, bf16* __restrict__ Xb,
                     int S, int D) {
  __shared__ float red[256];
  int row = blockIdx.x;           // b*S + s
  int s = row % S;
  int t = threadIdx.x;
  int id = ids[row];
  float v[3]; float sum = 0.f;
#pragma unroll
  for (int i = 0; i < 3; ++i) {
    int col = t + i * 256;
    float val = wemb[(size_t)id * D + col] + pemb[(size_t)s * D + col];
    v[i] = val; sum += val;
  }
  float mean = block_reduce_sum(sum, red) / (float)D;
  float vs = 0.f;
#pragma unroll
  for (int i = 0; i < 3; ++i) { float d = v[i] - mean; vs += d * d; }
  float var = block_reduce_sum(vs, red) / (float)D;
  float rstd = rsqrtf(var + 1e-5f);
#pragma unroll
  for (int i = 0; i < 3; ++i) {
    int col = t + i * 256;
    float y = (v[i] - mean) * rstd * g[col] + b[col];
    size_t off = (size_t)row * D + col;
    Xf[off] = y; Xb[off] = (bf16)y;
  }
}

// ---------------------------------------------------------------------------
// Residual add + LayerNorm (x = LN(x + h)); writes f32 + bf16
// ---------------------------------------------------------------------------
__global__ __launch_bounds__(256)
void add_ln_kernel(const float* __restrict__ X, const float* __restrict__ Hin,
                   const float* __restrict__ g, const float* __restrict__ b,
                   float* __restrict__ Xf, bf16* __restrict__ Xb, int D) {
  __shared__ float red[256];
  int row = blockIdx.x;
  int t = threadIdx.x;
  float v[3]; float sum = 0.f;
#pragma unroll
  for (int i = 0; i < 3; ++i) {
    int col = t + i * 256;
    size_t off = (size_t)row * D + col;
    float val = X[off] + Hin[off];
    v[i] = val; sum += val;
  }
  float mean = block_reduce_sum(sum, red) / (float)D;
  float vs = 0.f;
#pragma unroll
  for (int i = 0; i < 3; ++i) { float d = v[i] - mean; vs += d * d; }
  float var = block_reduce_sum(vs, red) / (float)D;
  float rstd = rsqrtf(var + 1e-5f);
#pragma unroll
  for (int i = 0; i < 3; ++i) {
    int col = t + i * 256;
    float y = (v[i] - mean) * rstd * g[col] + b[col];
    size_t off = (size_t)row * D + col;
    Xf[off] = y; Xb[off] = (bf16)y;
  }
}

// ---------------------------------------------------------------------------
// WMMA GEMM: C[M,N] = A[M,K] @ W[K,N] + bias ; optional GELU.
// Wt is the PRE-TRANSPOSED weight: Wt[n*K + k] = W[k*N + n].
// Tile 128x128, block = 256 threads = 8 waves (4x2), wave tile 32x64.
// Double-buffered LDS, async global->LDS staging. M,N % 128 == 0, K % 32 == 0.
// ---------------------------------------------------------------------------
template <int ACT>
__global__ __launch_bounds__(256)
void wmma_gemm_kernel(const bf16* __restrict__ A, const bf16* __restrict__ Wt,
                      const float* __restrict__ bias, float* __restrict__ Cf,
                      bf16* __restrict__ Cb, int M, int N, int K) {
  constexpr int TK = 32;
  __shared__ bf16 sA[2][128 * TK];   // [m][k]
  __shared__ bf16 sB[2][128 * TK];   // [n][k]
  int t = threadIdx.x;
  int lane = t & 31, wid = t >> 5;
  int m0 = blockIdx.y * 128, n0 = blockIdx.x * 128;
  int waveM = (wid >> 1) * 32, waveN = (wid & 1) * 64;

  // staging coordinates: 2 threads per 32-k row; each thread moves 2x16B
  int rowS = t >> 1, cg = (t & 1) * 16;
  const bf16* gA = A + (size_t)(m0 + rowS) * K + cg;
  const bf16* gB = Wt + (size_t)(n0 + rowS) * K + cg;
  int ldsOff = rowS * TK + cg;

  F8 acc[2][4];
#pragma unroll
  for (int rm = 0; rm < 2; ++rm)
#pragma unroll
    for (int cn = 0; cn < 4; ++cn)
#pragma unroll
      for (int r = 0; r < 8; ++r) acc[rm][cn].f[r] = 0.f;

  auto stage = [&](int buf, int k0) {
    async_copy16(gA + k0,     &sA[buf][ldsOff]);
    async_copy16(gA + k0 + 8, &sA[buf][ldsOff + 8]);
    async_copy16(gB + k0,     &sB[buf][ldsOff]);
    async_copy16(gB + k0 + 8, &sB[buf][ldsOff + 8]);
  };

  stage(0, 0);
  async_wait_all();
  __syncthreads();

  int cur = 0;
  for (int k0 = 0; k0 < K; k0 += TK) {
    bool more = (k0 + TK) < K;
    if (more) stage(cur ^ 1, k0 + TK);   // overlap next tile with compute

    const bf16* pA = sA[cur];
    const bf16* pB = sB[cur];
    v16bf fa[2], fb[4];
#pragma unroll
    for (int rm = 0; rm < 2; ++rm)
      fa[rm] = load_a_frag(pA + (waveM + rm * 16) * TK, TK);
#pragma unroll
    for (int cn = 0; cn < 4; ++cn)
      fb[cn] = load_bt_frag(pB + (waveN + cn * 16) * TK, TK);
#pragma unroll
    for (int rm = 0; rm < 2; ++rm)
#pragma unroll
      for (int cn = 0; cn < 4; ++cn)
        acc[rm][cn].v = wmma_bf16(fa[rm], fb[cn], acc[rm][cn].v);

    if (more) {
      async_wait_all();
      __syncthreads();
      cur ^= 1;
    }
  }

  // epilogue: C layout lane n=lane&15, VGPR r -> row r + 8*(lane>>4)
  int nIn = lane & 15, hi = lane >> 4;
#pragma unroll
  for (int rm = 0; rm < 2; ++rm)
#pragma unroll
    for (int cn = 0; cn < 4; ++cn) {
      int col = n0 + waveN + cn * 16 + nIn;
      float bv = bias ? bias[col] : 0.f;
#pragma unroll
      for (int r = 0; r < 8; ++r) {
        int row = m0 + waveM + rm * 16 + r + 8 * hi;
        float val = acc[rm][cn].f[r] + bv;
        if (ACT == 1) val = gelu_tanh(val);
        size_t off = (size_t)row * N + col;
        if (Cf) Cf[off] = val;
        if (Cb) Cb[off] = (bf16)val;
      }
    }
}

// ---------------------------------------------------------------------------
// Flash-style banded local attention + global column.
// grid = B*H*C*2 ; block = 256 (8 waves). Wave = 16 query rows.
// q/k/v: bf16 [B,S,D] with head h at columns h*64..h*64+63.
// ---------------------------------------------------------------------------
__global__ __launch_bounds__(256)
void local_attn_kernel(const bf16* __restrict__ Q, const bf16* __restrict__ Kc,
                       const bf16* __restrict__ Vc,
                       const int* __restrict__ amask, bf16* __restrict__ Out,
                       int S, int D, int Hh, int C, int Wwin) {
  const int Dh = 64;
  __shared__ bf16 sK[32 * 64];       // [key][dh]
  __shared__ bf16 sVt[64 * 32];      // [dh][key]
  __shared__ bf16 sP[8][16 * 32];    // per-wave P tile [m][key]
  __shared__ float sValid[32];

  int t = threadIdx.x, lane = t & 31, wid = t >> 5;
  int blk = blockIdx.x;
  int qhalf = blk & 1; blk >>= 1;
  int c = blk % C; blk /= C;
  int h = blk % Hh; int b = blk / Hh;
  int w = Wwin;
  int qoff = qhalf * 128 + wid * 16;          // query offset within chunk
  int m = lane & 15, hi = lane >> 4, n = lane & 15;
  int qrowAbs = c * w + qoff + m;
  const float scale = 0.125f;                  // 1/sqrt(64)

  // Q fragments (16 rows x 64 dh) straight from global
  v16bf qa0, qa1;
  {
    const bf16* qp = Q + ((size_t)b * S + qrowAbs) * D + h * Dh;
    bf16 e0[16], e1[16];
#pragma unroll
    for (int i = 0; i < 8; ++i) {
      e0[i]     = qp[hi * 8 + i];
      e0[8 + i] = qp[16 + hi * 8 + i];
      e1[i]     = qp[32 + hi * 8 + i];
      e1[8 + i] = qp[48 + hi * 8 + i];
    }
    __builtin_memcpy(&qa0, e0, sizeof(qa0));
    __builtin_memcpy(&qa1, e1, sizeof(qa1));
  }

  F8 o[4];
  float mrun[8], lrun[8];
#pragma unroll
  for (int cn = 0; cn < 4; ++cn)
#pragma unroll
    for (int r = 0; r < 8; ++r) o[cn].f[r] = 0.f;
#pragma unroll
  for (int r = 0; r < 8; ++r) { mrun[r] = -1e30f; lrun[r] = 0.f; }

  // block-level key-block range: queries [qmin, qmin+127] need j in [qmin, qmin+639]
  int qmin = qhalf * 128;
  int kb_lo = qmin >> 5;
  int kb_hi = (qmin + 127 + 2 * w) >> 5;
  int kbmax = (3 * w) / 32 - 1;
  if (kb_hi > kbmax) kb_hi = kbmax;

  for (int it = kb_lo; it <= kb_hi + 1; ++it) {
    int kb = it;
    bool gblk = (it == kb_hi + 1);  // the appended global-token column
    __syncthreads();
    // ---- stage 32 keys x 64 dh of K (async) and V transposed ----
    {
      int jj = t >> 3, dg = (t & 7) * 8;
      int tok = gblk ? (jj == 0 ? 0 : -1) : (c * w + kb * 32 + jj - w);
      bool ok = (tok >= 0) && (tok < S);
      if (dg == 0)
        sValid[jj] = (ok && amask[(size_t)b * S + (ok ? tok : 0)] > 0) ? 1.f : 0.f;
      uint4 vv = {0, 0, 0, 0};
      if (ok) {
        size_t base = ((size_t)b * S + tok) * D + h * Dh + dg;
        async_copy16(Kc + base, sK + jj * 64 + dg);
        vv = *reinterpret_cast<const uint4*>(Vc + base);
      } else {
        uint4 z = {0, 0, 0, 0};
        *reinterpret_cast<uint4*>(sK + jj * 64 + dg) = z;
      }
      bf16 tv[8]; __builtin_memcpy(tv, &vv, 16);
#pragma unroll
      for (int i = 0; i < 8; ++i) sVt[(dg + i) * 32 + jj] = tv[i];
    }
    async_wait_all();
    __syncthreads();

    // ---- scores: S[16x32] = Q(16x64) . K^T ----
    F8 st[2];
#pragma unroll
    for (int s2 = 0; s2 < 2; ++s2) {
#pragma unroll
      for (int r = 0; r < 8; ++r) st[s2].f[r] = 0.f;
      v16bf kb0 = load_bt_frag(sK + s2 * 16 * 64 + 0, 64);
      v16bf kb1 = load_bt_frag(sK + s2 * 16 * 64 + 32, 64);
      st[s2].v = wmma_bf16(qa0, kb0, st[s2].v);
      st[s2].v = wmma_bf16(qa1, kb1, st[s2].v);
    }

    // ---- mask + scale + online softmax ----
#pragma unroll
    for (int r = 0; r < 8; ++r) {
      int mrow = r + 8 * hi;          // query row in wave tile
      int qi = qoff + mrow;           // query index within chunk
#pragma unroll
      for (int s2 = 0; s2 < 2; ++s2) {
        int jj = s2 * 16 + n;
        int j = kb * 32 + jj;
        int dlt = j - w - qi;
        bool band = gblk ? (jj == 0) : (dlt <= w && dlt >= -w);
        bool ok = band && (sValid[jj] > 0.5f);
        st[s2].f[r] = ok ? st[s2].f[r] * scale : -1e9f;
      }
      float rm = fmaxf(st[0].f[r], st[1].f[r]);
#pragma unroll
      for (int off = 1; off < 16; off <<= 1) rm = fmaxf(rm, __shfl_xor(rm, off));
      float mnew = fmaxf(mrun[r], rm);
      float alpha = __expf(mrun[r] - mnew);
      float p0 = __expf(st[0].f[r] - mnew);
      float p1 = __expf(st[1].f[r] - mnew);
      float rs = p0 + p1;
#pragma unroll
      for (int off = 1; off < 16; off <<= 1) rs += __shfl_xor(rs, off);
      lrun[r] = lrun[r] * alpha + rs;
      mrun[r] = mnew;
#pragma unroll
      for (int cn = 0; cn < 4; ++cn) o[cn].f[r] *= alpha;
      sP[wid][mrow * 32 + n]      = (bf16)p0;
      sP[wid][mrow * 32 + 16 + n] = (bf16)p1;
    }
    __syncthreads();  // P tiles visible (D-layout -> A-layout redistribution)

    // ---- O += P(16x32) . V(32x64) ----
    v16bf pa = load_a_frag(&sP[wid][0], 32);
#pragma unroll
    for (int cn = 0; cn < 4; ++cn) {
      v16bf vb = load_bt_frag(sVt + cn * 16 * 32, 32);
      o[cn].v = wmma_bf16(pa, vb, o[cn].v);
    }
  }

  // ---- normalize + store ----
#pragma unroll
  for (int r = 0; r < 8; ++r) {
    float inv = lrun[r] > 0.f ? 1.f / lrun[r] : 0.f;
    int qabs = c * w + qoff + r + 8 * hi;
#pragma unroll
    for (int cn = 0; cn < 4; ++cn) {
      int dh = cn * 16 + n;
      Out[((size_t)b * S + qabs) * D + h * Dh + dh] = (bf16)(o[cn].f[r] * inv);
    }
  }
}

// ---------------------------------------------------------------------------
// Global (CLS) full attention with the separate global projections.
// grid = B*H ; block = 64 threads. Overwrites Out row s=0.
// ---------------------------------------------------------------------------
__global__ __launch_bounds__(64)
void global_attn_kernel(const bf16* __restrict__ Qg, const bf16* __restrict__ Kg,
                        const bf16* __restrict__ Vg,
                        const int* __restrict__ amask, bf16* __restrict__ Out,
                        int S, int D, int Hh) {
  __shared__ float sc[4096];
  __shared__ float red[64];
  const int Dh = 64;
  int bh = blockIdx.x; int h = bh % Hh; int b = bh / Hh;
  int t = threadIdx.x;
  const bf16* qg = Qg + (size_t)b * S * D + h * Dh;
  const float scale = 0.125f;
  for (int s = t; s < S; s += 64) {
    const bf16* kp = Kg + ((size_t)b * S + s) * D + h * Dh;
    float acc = 0.f;
#pragma unroll 8
    for (int d = 0; d < Dh; ++d) acc += (float)qg[d] * (float)kp[d];
    sc[s] = (amask[(size_t)b * S + s] > 0) ? acc * scale : -1e9f;
  }
  __syncthreads();
  float mx = -1e30f;
  for (int s = t; s < S; s += 64) mx = fmaxf(mx, sc[s]);
  red[t] = mx; __syncthreads();
  for (int o = 32; o > 0; o >>= 1) {
    if (t < o) red[t] = fmaxf(red[t], red[t + o]);
    __syncthreads();
  }
  mx = red[0]; __syncthreads();
  float sum = 0.f;
  for (int s = t; s < S; s += 64) {
    float p = __expf(sc[s] - mx);
    sc[s] = p; sum += p;
  }
  __syncthreads();
  red[t] = sum; __syncthreads();
  for (int o = 32; o > 0; o >>= 1) {
    if (t < o) red[t] += red[t + o];
    __syncthreads();
  }
  float inv = 1.f / red[0];
  float acc = 0.f;
  for (int s = 0; s < S; ++s)
    acc += sc[s] * (float)Vg[((size_t)b * S + s) * D + h * Dh + t];
  Out[(size_t)b * S * D + h * Dh + t] = (bf16)(acc * inv);
}

// ---------------------------------------------------------------------------
// Pooler + classifier: out[b,:] = tanh(x[b,0] @ Wp + bp) @ Wc + bc
// ---------------------------------------------------------------------------
__global__ __launch_bounds__(256)
void pooler_kernel(const float* __restrict__ X, const float* __restrict__ Wp,
                   const float* __restrict__ bp, const float* __restrict__ Wc,
                   const float* __restrict__ bc, float* __restrict__ out,
                   int S, int D) {
  __shared__ float pooled[768];
  __shared__ float red[256];
  int b = blockIdx.x, t = threadIdx.x;
  const float* x0 = X + (size_t)b * S * D;
#pragma unroll
  for (int i = 0; i < 3; ++i) {
    int d = t + i * 256;
    float acc = bp[d];
    for (int k = 0; k < D; ++k) acc += x0[k] * Wp[(size_t)k * D + d];
    pooled[d] = tanhf(acc);
  }
  __syncthreads();
  for (int c2 = 0; c2 < 2; ++c2) {
    float partial = 0.f;
#pragma unroll
    for (int i = 0; i < 3; ++i) {
      int d = t + i * 256;
      partial += pooled[d] * Wc[(size_t)d * 2 + c2];
    }
    float ssum = block_reduce_sum(partial, red);
    if (t == 0) out[b * 2 + c2] = ssum + bc[c2];
  }
}

// ---------------------------------------------------------------------------
// Host orchestration
// ---------------------------------------------------------------------------
static inline void launch_gemm(const bf16* A, const bf16* Wt, const float* bias,
                               float* Cf, bf16* Cb, int M, int N, int K, int act,
                               hipStream_t st) {
  dim3 g(N / 128, M / 128), blk(256);
  if (act)
    wmma_gemm_kernel<1><<<g, blk, 0, st>>>(A, Wt, bias, Cf, Cb, M, N, K);
  else
    wmma_gemm_kernel<0><<<g, blk, 0, st>>>(A, Wt, bias, Cf, Cb, M, N, K);
}

extern "C" void kernel_launch(void* const* d_in, const int* in_sizes, int n_in,
                              void* d_out, int out_size, void* d_ws,
                              size_t ws_size, hipStream_t stream) {
  (void)in_sizes; (void)n_in; (void)out_size; (void)ws_size;
  const int B = 2, S = 4096, D = 768, Hh = 12, F = 3072, NL = 2, Wwin = 256;
  const int C = S / Wwin;
  const int M = B * S;

  const int*   input_ids = (const int*)d_in[0];
  const int*   amask     = (const int*)d_in[1];
  const float* word_emb  = (const float*)d_in[2];
  const float* pos_emb   = (const float*)d_in[3];
  const float* ln_e_g    = (const float*)d_in[4];
  const float* ln_e_b    = (const float*)d_in[5];
  const float* Wq  = (const float*)d_in[6];   const float* bq  = (const float*)d_in[7];
  const float* Wk  = (const float*)d_in[8];   const float* bk  = (const float*)d_in[9];
  const float* Wv  = (const float*)d_in[10];  const float* bv  = (const float*)d_in[11];
  const float* Wqg = (const float*)d_in[12];  const float* bqg = (const float*)d_in[13];
  const float* Wkg = (const float*)d_in[14];  const float* bkg = (const float*)d_in[15];
  const float* Wvg = (const float*)d_in[16];  const float* bvg = (const float*)d_in[17];
  const float* Wo  = (const float*)d_in[18];  const float* bo  = (const float*)d_in[19];
  const float* ln1_g = (const float*)d_in[20]; const float* ln1_b = (const float*)d_in[21];
  const float* W1  = (const float*)d_in[22];  const float* bf1 = (const float*)d_in[23];
  const float* W2  = (const float*)d_in[24];  const float* bf2 = (const float*)d_in[25];
  const float* ln2_g = (const float*)d_in[26]; const float* ln2_b = (const float*)d_in[27];
  const float* Wp  = (const float*)d_in[28];  const float* bp  = (const float*)d_in[29];
  const float* Wc  = (const float*)d_in[30];  const float* bc  = (const float*)d_in[31];

  // ---- workspace allocator ----
  char* base = (char*)d_ws; size_t off = 0;
  auto alloc = [&](size_t bytes) -> void* {
    void* p = base + off;
    off = (off + bytes + 255) & ~(size_t)255;
    return p;
  };
  const size_t WDD = (size_t)D * D, WDF = (size_t)D * F;
  bf16* WqB  = (bf16*)alloc((size_t)NL * WDD * 2);
  bf16* WkB  = (bf16*)alloc((size_t)NL * WDD * 2);
  bf16* WvB  = (bf16*)alloc((size_t)NL * WDD * 2);
  bf16* WqgB = (bf16*)alloc((size_t)NL * WDD * 2);
  bf16* WkgB = (bf16*)alloc((size_t)NL * WDD * 2);
  bf16* WvgB = (bf16*)alloc((size_t)NL * WDD * 2);
  bf16* WoB  = (bf16*)alloc((size_t)NL * WDD * 2);
  bf16* W1B  = (bf16*)alloc((size_t)NL * WDF * 2);
  bf16* W2B  = (bf16*)alloc((size_t)NL * WDF * 2);
  float* xf  = (float*)alloc((size_t)M * D * 4);
  bf16*  xb  = (bf16*)alloc((size_t)M * D * 2);
  bf16*  qb  = (bf16*)alloc((size_t)M * D * 2);
  bf16*  kbf = (bf16*)alloc((size_t)M * D * 2);
  bf16*  vbf = (bf16*)alloc((size_t)M * D * 2);
  bf16*  qgb = (bf16*)alloc((size_t)M * D * 2);
  bf16*  kgb = (bf16*)alloc((size_t)M * D * 2);
  bf16*  vgb = (bf16*)alloc((size_t)M * D * 2);
  bf16*  aob = (bf16*)alloc((size_t)M * D * 2);
  float* hf  = (float*)alloc((size_t)M * D * 4);
  bf16*  midb = (bf16*)alloc((size_t)M * F * 2);

  // ---- convert + transpose weights to bf16 [N][K] ----
  auto cvtT = [&](const float* src, bf16* dst, int K, int N) {
    dim3 g(N / 32, K / 32), blk(256);
    cvt_transpose_kernel<<<g, blk, 0, stream>>>(src, dst, K, N);
  };
  for (int l = 0; l < NL; ++l) {
    cvtT(Wq  + (size_t)l * WDD, WqB  + (size_t)l * WDD, D, D);
    cvtT(Wk  + (size_t)l * WDD, WkB  + (size_t)l * WDD, D, D);
    cvtT(Wv  + (size_t)l * WDD, WvB  + (size_t)l * WDD, D, D);
    cvtT(Wqg + (size_t)l * WDD, WqgB + (size_t)l * WDD, D, D);
    cvtT(Wkg + (size_t)l * WDD, WkgB + (size_t)l * WDD, D, D);
    cvtT(Wvg + (size_t)l * WDD, WvgB + (size_t)l * WDD, D, D);
    cvtT(Wo  + (size_t)l * WDD, WoB  + (size_t)l * WDD, D, D);
    cvtT(W1  + (size_t)l * WDF, W1B  + (size_t)l * WDF, D, F);  // [F][D]
    cvtT(W2  + (size_t)l * WDF, W2B  + (size_t)l * WDF, F, D);  // [D][F]
  }

  // ---- embeddings + LN ----
  embed_ln_kernel<<<M, 256, 0, stream>>>(input_ids, word_emb, pos_emb, ln_e_g,
                                         ln_e_b, xf, xb, S, D);

  for (int l = 0; l < NL; ++l) {
    // projections (bf16 WMMA)
    launch_gemm(xb, WqB  + (size_t)l * WDD, bq  + (size_t)l * D, nullptr, qb,  M, D, D, 0, stream);
    launch_gemm(xb, WkB  + (size_t)l * WDD, bk  + (size_t)l * D, nullptr, kbf, M, D, D, 0, stream);
    launch_gemm(xb, WvB  + (size_t)l * WDD, bv  + (size_t)l * D, nullptr, vbf, M, D, D, 0, stream);
    launch_gemm(xb, WqgB + (size_t)l * WDD, bqg + (size_t)l * D, nullptr, qgb, M, D, D, 0, stream);
    launch_gemm(xb, WkgB + (size_t)l * WDD, bkg + (size_t)l * D, nullptr, kgb, M, D, D, 0, stream);
    launch_gemm(xb, WvgB + (size_t)l * WDD, bvg + (size_t)l * D, nullptr, vgb, M, D, D, 0, stream);

    // banded local attention (flash-style, WMMA) + CLS global attention
    local_attn_kernel<<<B * Hh * C * 2, 256, 0, stream>>>(qb, kbf, vbf, amask,
                                                          aob, S, D, Hh, C, Wwin);
    global_attn_kernel<<<B * Hh, 64, 0, stream>>>(qgb, kgb, vgb, amask, aob,
                                                  S, D, Hh);

    // output projection + residual LN
    launch_gemm(aob, WoB + (size_t)l * WDD, bo + (size_t)l * D, hf, nullptr, M, D, D, 0, stream);
    add_ln_kernel<<<M, 256, 0, stream>>>(xf, hf, ln1_g + (size_t)l * D,
                                         ln1_b + (size_t)l * D, xf, xb, D);

    // FFN
    launch_gemm(xb, W1B + (size_t)l * WDF, bf1 + (size_t)l * F, nullptr, midb, M, F, D, 1, stream);
    launch_gemm(midb, W2B + (size_t)l * WDF, bf2 + (size_t)l * D, hf, nullptr, M, D, F, 0, stream);
    add_ln_kernel<<<M, 256, 0, stream>>>(xf, hf, ln2_g + (size_t)l * D,
                                         ln2_b + (size_t)l * D, xf, xb, D);
  }

  // ---- pooler + classifier ----
  pooler_kernel<<<B, 256, 0, stream>>>(xf, Wp, bp, Wc, bc, (float*)d_out, S, D);
}
